// DotProductAttention_11467562680789
// MI455X (gfx1250) — matrix operand order repaired and checked by reading, not compiled, not run on hardware
//
#include <hip/hip_runtime.h>
#include <hip/hip_bf16.h>

#define B_   16
#define LQ_  2048
#define LK_  2048
#define D_   128
#define NT64_ 32                // LK_/64 key tiles per batch
#define TILE_USHORT 8192        // 64x128 bf16 elements per tile (16 KB)

typedef __attribute__((ext_vector_type(16))) __bf16 v16bf;
typedef __attribute__((ext_vector_type(8)))  float  v8f;

// 16 bf16 = 32 B = 8 VGPRs/lane (A or B fragment of v_wmma_f32_16x16x32_bf16, wave32)
union Frag16 {
  unsigned short u[16];
  uint4          q[2];
  v16bf          v;
};

__device__ __forceinline__ unsigned short f2bf(float f) {
  unsigned int u = __builtin_bit_cast(unsigned int, f);
  u += 0x7fffu + ((u >> 16) & 1u);     // round-to-nearest-even
  return (unsigned short)(u >> 16);
}

// generic pointer -> 32-bit LDS byte offset
__device__ __forceinline__ unsigned lds_off_of(const void* p) {
  return (unsigned)(unsigned long long)
         (__attribute__((address_space(3))) const void*)p;
}

// per-lane 16 B global -> LDS, ASYNCcnt-tracked (CDNA5 async copy, GVS mode)
__device__ __forceinline__ void async_ld16(unsigned lds_byte, unsigned goff,
                                           unsigned long long gbase) {
  asm volatile("global_load_async_to_lds_b128 %0, %1, %2"
               :: "v"(lds_byte), "v"(goff), "s"(gbase) : "memory");
}

// max-reduction across the 16-lane half: VALU permlane butterflies (XOR patterns)
__device__ __forceinline__ float xmax16(float x) {
#if __has_builtin(__builtin_amdgcn_permlane16)
#define PL16_(v, lo, hi)                                                   \
  __builtin_bit_cast(float, __builtin_amdgcn_permlane16(                   \
      __builtin_bit_cast(unsigned, v), __builtin_bit_cast(unsigned, v),    \
      (lo), (hi), false, false))
  x = fmaxf(x, PL16_(x, 0x67452301u, 0xEFCDAB89u));   // xor 1
  x = fmaxf(x, PL16_(x, 0x54761032u, 0xDCFE98BAu));   // xor 2
  x = fmaxf(x, PL16_(x, 0x32107654u, 0xBA98FEDCu));   // xor 4
  x = fmaxf(x, PL16_(x, 0xFEDCBA98u, 0x76543210u));   // xor 8
#undef PL16_
#else
  x = fmaxf(x, __shfl_xor(x, 1));
  x = fmaxf(x, __shfl_xor(x, 2));
  x = fmaxf(x, __shfl_xor(x, 4));
  x = fmaxf(x, __shfl_xor(x, 8));
#endif
  return x;
}

// (1/sqrt(128)) * log2(e) : softmax in exp2 domain
#define KSC 0.12751741f

// ---------------------------------------------------------------------------
// Pre-pass: K,V f32 -> bf16 once, 16 KB tile-contiguous blocks in workspace.
//   KtWs[b][t] : 64 keys x 128 d, row-major
//   VtWs[b][t] : 128 d x 64 keys (transposed -> contiguous B-fragments)
// ---------------------------------------------------------------------------
__launch_bounds__(256, 2)
__global__ void cvt_kv_kernel(const float* __restrict__ Kg,
                              const float* __restrict__ Vg,
                              unsigned short* __restrict__ KtWs,
                              unsigned short* __restrict__ VtWs)
{
  __shared__ __align__(16) unsigned short tr[128 * 64];   // 16 KB
  const int t = blockIdx.x, b = blockIdx.y, tid = threadIdx.x;
  const size_t gsrc = ((size_t)b * LK_ + t * 64) * D_;
  const size_t gdst = (size_t)(b * NT64_ + t) * TILE_USHORT;

#pragma unroll
  for (int i = 0; i < 8; ++i) {
    const int idx4 = tid + i * 256;          // float4 slot in 64x128 tile
    const int row  = idx4 >> 5;
    const int c4   = (idx4 & 31) << 2;
    const float4 kv = *(const float4*)(Kg + gsrc + row * D_ + c4);
    uint2 pk;
    pk.x = (unsigned)f2bf(kv.x) | ((unsigned)f2bf(kv.y) << 16);
    pk.y = (unsigned)f2bf(kv.z) | ((unsigned)f2bf(kv.w) << 16);
    *(uint2*)&KtWs[gdst + row * 128 + c4] = pk;
    const float4 vv = *(const float4*)(Vg + gsrc + row * D_ + c4);
    tr[(c4 + 0) * 64 + row] = f2bf(vv.x);    // transpose through LDS
    tr[(c4 + 1) * 64 + row] = f2bf(vv.y);
    tr[(c4 + 2) * 64 + row] = f2bf(vv.z);
    tr[(c4 + 3) * 64 + row] = f2bf(vv.w);
  }
  __syncthreads();
#pragma unroll
  for (int i = 0; i < 4; ++i) {              // contiguous b128 stores of V^T
    const int off = (tid + i * 256) * 8;
    *(uint4*)&VtWs[gdst + off] = *(const uint4*)&tr[off];
  }
}

// ---------------------------------------------------------------------------
// Flash attention forward: wave = 16 query rows, block = 8 waves = 128 rows.
// 64-key tiles, double-buffered async staging, multi-buffered LDS fragments,
// WMMA row-sums, mask-free interior softmax.
// ---------------------------------------------------------------------------
__launch_bounds__(256, 2)
__global__ void fa_fwd_kernel(const float* __restrict__ Qg,
                              const unsigned short* __restrict__ KtWs,
                              const unsigned short* __restrict__ VtWs,
                              const long long* __restrict__ VL,
                              float* __restrict__ Out)
{
  __shared__ __align__(16) unsigned short Kbuf[2][TILE_USHORT]; // 2 x 16 KB
  __shared__ __align__(16) unsigned short Vbuf[2][TILE_USHORT]; // 2 x 16 KB
  __shared__ __align__(16) unsigned short Plds[8 * 16 * 64];    // 16 KB

  const int tid  = threadIdx.x;
  const int wave = tid >> 5;
  const int lane = tid & 31;
  const int half = lane >> 4;
  const int lc   = lane & 15;

  const int b  = blockIdx.y;
  const int q0 = blockIdx.x * 128 + wave * 16;

  const long long vl = VL[b];
  const int nkt = (int)((vl + 63) >> 6);     // 64-key tiles with >=1 valid key

  const unsigned kb0 = lds_off_of(&Kbuf[0][0]);
  const unsigned vb0 = lds_off_of(&Vbuf[0][0]);
  const unsigned coff = (unsigned)(wave * 128 + lane) * 16u;

  // ---- Q fragments (A layout), loaded once -------------------------------
  Frag16 qf[4];
  {
    const float* qrow = Qg + ((size_t)b * LQ_ + q0 + lc) * D_;
    const int kbase = half * 16;
#pragma unroll
    for (int c = 0; c < 4; ++c)
#pragma unroll
      for (int e = 0; e < 16; ++e)
        qf[c].u[e] = f2bf(qrow[c * 32 + kbase + e]);
  }

  Frag16 ones;                               // all-ones B fragment for row-sums
#pragma unroll
  for (int e = 0; e < 16; ++e) ones.u[e] = 0x3F80u;   // bf16 1.0

  v8f acc[8];
#pragma unroll
  for (int t = 0; t < 8; ++t) acc[t] = (v8f)(0.0f);
  v8f lacc = (v8f)(0.0f);                    // row-sums (C layout, replicated)
  float mrow[8];
#pragma unroll
  for (int r = 0; r < 8; ++r) mrow[r] = -1e30f;

  const unsigned long long kws = (unsigned long long)(KtWs + (size_t)b * NT64_ * TILE_USHORT);
  const unsigned long long vws = (unsigned long long)(VtWs + (size_t)b * NT64_ * TILE_USHORT);

  // prologue: async-stage tile 0 into buffer 0 (8 issues per wave)
#pragma unroll
  for (int i = 0; i < 4; ++i) {
    async_ld16(kb0 + coff + i * 512, coff + i * 512, kws);
    async_ld16(vb0 + coff + i * 512, coff + i * 512, vws);
  }

  for (int kt = 0; kt < nkt; ++kt) {
    const int ktb = kt * 64;

    if (kt + 1 < nkt) {   // stage next tile into the other buffer
      const unsigned nk = kb0 + (unsigned)((kt + 1) & 1) * (TILE_USHORT * 2);
      const unsigned nv = vb0 + (unsigned)((kt + 1) & 1) * (TILE_USHORT * 2);
      const unsigned long long kbase = kws + (unsigned long long)(kt + 1) * (TILE_USHORT * 2);
      const unsigned long long vbase = vws + (unsigned long long)(kt + 1) * (TILE_USHORT * 2);
#pragma unroll
      for (int i = 0; i < 4; ++i) {
        async_ld16(nk + coff + i * 512, coff + i * 512, kbase);
        async_ld16(nv + coff + i * 512, coff + i * 512, vbase);
      }
      // async loads complete in order: <=8 outstanding => current tile landed
      asm volatile("s_wait_asynccnt 0x8" ::: "memory");
    } else {
      asm volatile("s_wait_asynccnt 0x0" ::: "memory");
    }
    __syncthreads();      // all waves' portions of the current tile visible

    const unsigned short* Kl = (const unsigned short*)&Kbuf[kt & 1][0];
    const unsigned short* Vl = (const unsigned short*)&Vbuf[kt & 1][0];

    // -------- S = Q * K^T : four 16x16 f32 tiles over 64 keys -------------
    // double-buffered B-fragments: loads of chunk c+1 overlap WMMAs of chunk c
    v8f S[4];
#pragma unroll
    for (int j = 0; j < 4; ++j) S[j] = (v8f)(0.0f);
    Frag16 kf[2][4];
#pragma unroll
    for (int j = 0; j < 4; ++j) {
      const uint4* p = (const uint4*)&Kl[(j * 16 + lc) * 128 + half * 16];
      kf[0][j].q[0] = p[0]; kf[0][j].q[1] = p[1];
    }
#pragma unroll
    for (int c = 0; c < 4; ++c) {
      if (c < 3) {
#pragma unroll
        for (int j = 0; j < 4; ++j) {
          const uint4* p = (const uint4*)&Kl[(j * 16 + lc) * 128 + (c + 1) * 32 + half * 16];
          kf[(c + 1) & 1][j].q[0] = p[0];
          kf[(c + 1) & 1][j].q[1] = p[1];
        }
      }
#pragma unroll
      for (int j = 0; j < 4; ++j)
        S[j] = __builtin_amdgcn_wmma_f32_16x16x32_bf16(false, qf[c].v, false, kf[c & 1][j].v,
                                                       (short)0, S[j], false, false);
    }

    // -------- online softmax ----------------------------------------------
    if (ktb + 64 <= (int)vl) {
      // interior tile: no masking needed
#pragma unroll
      for (int r = 0; r < 8; ++r) {
        float s[4];
#pragma unroll
        for (int j = 0; j < 4; ++j) s[j] = S[j][r] * KSC;
        float mx = fmaxf(fmaxf(s[0], s[1]), fmaxf(s[2], s[3]));
        mx = xmax16(mx);
        const float mnew = fmaxf(mrow[r], mx);
        const float corr = exp2f(mrow[r] - mnew);
        mrow[r] = mnew;
        lacc[r] *= corr;
#pragma unroll
        for (int t = 0; t < 8; ++t) acc[t][r] *= corr;
        const int prow = half * 8 + r;
#pragma unroll
        for (int j = 0; j < 4; ++j)
          Plds[wave * 1024 + prow * 64 + j * 16 + lc] = f2bf(exp2f(s[j] - mnew));
      }
    } else {
      // tail tile straddling valid_len: mask keys >= vl
      bool ok[4];
#pragma unroll
      for (int j = 0; j < 4; ++j) ok[j] = (long long)(ktb + j * 16 + lc) < vl;
#pragma unroll
      for (int r = 0; r < 8; ++r) {
        float s[4];
#pragma unroll
        for (int j = 0; j < 4; ++j) s[j] = ok[j] ? S[j][r] * KSC : -1e30f;
        float mx = fmaxf(fmaxf(s[0], s[1]), fmaxf(s[2], s[3]));
        mx = xmax16(mx);
        const float mnew = fmaxf(mrow[r], mx);
        const float corr = exp2f(mrow[r] - mnew);
        mrow[r] = mnew;
        lacc[r] *= corr;
#pragma unroll
        for (int t = 0; t < 8; ++t) acc[t][r] *= corr;
        const int prow = half * 8 + r;
#pragma unroll
        for (int j = 0; j < 4; ++j)
          Plds[wave * 1024 + prow * 64 + j * 16 + lc] = f2bf(exp2f(s[j] - mnew));
      }
    }

    asm volatile("s_wait_dscnt 0" ::: "memory"); // wave-local P RAW

    // -------- P fragments (two 16x32 A tiles over 64 keys) ----------------
    Frag16 pf[2];
    {
      const int kbase = half * 16;
#pragma unroll
      for (int ka = 0; ka < 2; ++ka) {
        const uint4* p0 = (const uint4*)&Plds[wave * 1024 + lc * 64 + ka * 32 + kbase];
        const uint4* p1 = (const uint4*)&Plds[wave * 1024 + lc * 64 + ka * 32 + kbase + 8];
        pf[ka].q[0] = p0[0];
        pf[ka].q[1] = p1[0];
      }
    }

    // row-sums via WMMA against the all-ones B fragment (no lane shuffles)
    lacc = __builtin_amdgcn_wmma_f32_16x16x32_bf16(false, pf[0].v, false, ones.v,
                                                   (short)0, lacc, false, false);
    lacc = __builtin_amdgcn_wmma_f32_16x16x32_bf16(false, pf[1].v, false, ones.v,
                                                   (short)0, lacc, false, false);

    // -------- O += P * V (double-buffered V fragments) --------------------
    Frag16 vf[2][2];
#pragma unroll
    for (int ka = 0; ka < 2; ++ka) {
      const uint4* pv = (const uint4*)&Vl[lc * 64 + ka * 32 + half * 16];
      vf[0][ka].q[0] = pv[0]; vf[0][ka].q[1] = pv[1];
    }
#pragma unroll
    for (int t = 0; t < 8; ++t) {
      if (t < 7) {
#pragma unroll
        for (int ka = 0; ka < 2; ++ka) {
          const uint4* pv = (const uint4*)&Vl[((t + 1) * 16 + lc) * 64 + ka * 32 + half * 16];
          vf[(t + 1) & 1][ka].q[0] = pv[0];
          vf[(t + 1) & 1][ka].q[1] = pv[1];
        }
      }
#pragma unroll
      for (int ka = 0; ka < 2; ++ka)
        acc[t] = __builtin_amdgcn_wmma_f32_16x16x32_bf16(false, pf[ka].v, false, vf[t & 1][ka].v,
                                                         (short)0, acc[t], false, false);
    }

    __syncthreads();    // all reads of this buffer done before it is re-staged
  }

  // -------- epilogue: O = acc / rowsum ------------------------------------
#pragma unroll
  for (int r = 0; r < 8; ++r) {
    const float inv = 1.0f / lacc[r];
    float* orow = Out + ((size_t)b * LQ_ + q0 + half * 8 + r) * D_;
#pragma unroll
    for (int t = 0; t < 8; ++t) orow[t * 16 + lc] = acc[t][r] * inv;
  }
}

extern "C" void kernel_launch(void* const* d_in, const int* in_sizes, int n_in,
                              void* d_out, int out_size, void* d_ws, size_t ws_size,
                              hipStream_t stream) {
  const float*     Q  = (const float*)d_in[0];
  const float*     K  = (const float*)d_in[1];
  const float*     V  = (const float*)d_in[2];
  const long long* vl = (const long long*)d_in[3];
  float* out = (float*)d_out;
  (void)in_sizes; (void)n_in; (void)out_size; (void)ws_size;

  unsigned short* KtWs = (unsigned short*)d_ws;                    // 8 MB
  unsigned short* VtWs = KtWs + (size_t)B_ * NT64_ * TILE_USHORT;  // 8 MB

  cvt_kv_kernel<<<dim3(NT64_, B_), 256, 0, stream>>>(K, V, KtWs, VtWs);
  fa_fwd_kernel<<<dim3(LQ_ / 128, B_), 256, 0, stream>>>(Q, KtWs, VtWs, vl, out);
}
